// EliteTopoBrain_18769007083997
// MI455X (gfx1250) — matrix operand order repaired and checked by reading, not compiled, not run on hardware
//
#include <hip/hip_runtime.h>
#include <hip/hip_bf16.h>
#include <math.h>

typedef __attribute__((ext_vector_type(16))) __bf16 bf16x16;
typedef __attribute__((ext_vector_type(8)))  __bf16 bf16x8;
typedef __attribute__((ext_vector_type(4)))  __bf16 bf16x4;
typedef __attribute__((ext_vector_type(8)))  float  f32x8;

#define BSZ   4096
#define FDIM  512
#define HDIM  1024
#define NN    64
#define DD    64
#define FLATD 4096
#define CCLS  1000
#define CAPM  64

// lane stride (in bf16 elems) for fragment-ordered LDS: 16 data + 8 pad
#define FRAG_STRIDE 24

// ---------------------------------------------------------------------------
// bf16 WMMA GEMM:  C[z] = act( A[z] @ B[z]^T + bias[z] (+ C_old) )
// A bf16 [M,K] row-major; B bf16 [Nc,K] row-major (always A@B^T).
// Block tile 128(M) x 128(N); 8 waves (2x4); each wave 64x32 via 4x2 WMMA.
// LDS tiles are stored in WMMA *fragment order*; with bf16 sources each
// 8-element k-chunk is a contiguous 16B run, so staging is pure
// global_load_b128 -> ds_store_b128 and fragment reads are 2x ds_load_b128.
// ---------------------------------------------------------------------------
__global__ __launch_bounds__(256) void gemm_bf16_wmma(
    const __bf16* __restrict__ A, long long lda, long long sA,
    const __bf16* __restrict__ Bm, long long ldb, long long sB,
    const float* __restrict__ bias, long long sBias,
    float* __restrict__ C, long long ldc, long long sC,
    int M, int Nc, int K, int accumulate, int act, int outBf16)
{
  __shared__ __bf16 Afr[8 * 32 * FRAG_STRIDE];   // 8 M-subtiles
  __shared__ __bf16 Bfr[8 * 32 * FRAG_STRIDE];   // 8 N-subtiles

  const int tid  = threadIdx.x;
  const int lane = tid & 31;
  const int wid  = tid >> 5;
  const int wm   = wid >> 2;            // 0..1  (M direction, 64 rows each)
  const int wn   = wid & 3;             // 0..3  (N direction, 32 cols each)

  const long long z = blockIdx.z;
  const __bf16* Ab = A  + z * sA;
  const __bf16* Bb = Bm + z * sB;
  float*        Cb = C  + z * sC;
  __bf16*       Cb16 = (__bf16*)C + z * sC;
  const int bm = blockIdx.x * 128;
  const int bn = blockIdx.y * 128;

  f32x8 acc[4][2] = {};

  for (int k0 = 0; k0 < K; k0 += 32) {
    // ---- stage A tile 128x32: 16B chunks straight into fragment order ----
#pragma unroll
    for (int i = 0; i < 2; ++i) {
      int q = tid + i * 256;            // chunk 0..511
      int r  = q >> 2;                  // 0..127
      int cc = (q & 3) << 3;            // k-chunk start: 0,8,16,24
      bf16x8 vA = *(const bf16x8*)(Ab + (long long)(bm + r) * lda + (k0 + cc));
      int slot = (((cc >> 3) & 1) << 4) + (r & 15);
      *(bf16x8*)(&Afr[((r >> 4) * 32 + slot) * FRAG_STRIDE + ((cc & 16) >> 1)]) = vA;
    }
    // ---- stage B tile 128(n) x 32(k) ----
#pragma unroll
    for (int i = 0; i < 2; ++i) {
      int q = tid + i * 256;
      int n  = q >> 2;
      int cc = (q & 3) << 3;
      bf16x8 vB;
      if (bn + n < Nc) {
        vB = *(const bf16x8*)(Bb + (long long)(bn + n) * ldb + (k0 + cc));
      } else {
#pragma unroll
        for (int e = 0; e < 8; ++e) vB[e] = (__bf16)0.0f;
      }
      int slot = (((cc >> 3) & 1) << 4) + (n & 15);
      *(bf16x8*)(&Bfr[((n >> 4) * 32 + slot) * FRAG_STRIDE + ((cc & 16) >> 1)]) = vB;
    }
    if (k0 + 32 < K)   // global_prefetch_b8 for next A tile
      __builtin_prefetch(Ab + (long long)(bm + (tid & 127)) * lda + (k0 + 32), 0, 1);
    __syncthreads();

    // ---- fragments: 32B contiguous per fragment (2x ds_load_b128) ----
    bf16x16 afrag[4], bfrag[2];
#pragma unroll
    for (int mi = 0; mi < 4; ++mi) {
      const __bf16* pa = &Afr[((wm * 4 + mi) * 32 + lane) * FRAG_STRIDE];
      bf16x8 lo = *(const bf16x8*)pa;
      bf16x8 hi = *(const bf16x8*)(pa + 8);
      afrag[mi] = __builtin_shufflevector(lo, hi, 0,1,2,3,4,5,6,7,8,9,10,11,12,13,14,15);
    }
#pragma unroll
    for (int ni = 0; ni < 2; ++ni) {
      const __bf16* pb = &Bfr[((wn * 2 + ni) * 32 + lane) * FRAG_STRIDE];
      bf16x8 lo = *(const bf16x8*)pb;
      bf16x8 hi = *(const bf16x8*)(pb + 8);
      bfrag[ni] = __builtin_shufflevector(lo, hi, 0,1,2,3,4,5,6,7,8,9,10,11,12,13,14,15);
    }
#pragma unroll
    for (int mi = 0; mi < 4; ++mi)
#pragma unroll
      for (int ni = 0; ni < 2; ++ni)
        acc[mi][ni] = __builtin_amdgcn_wmma_f32_16x16x32_bf16(
            false, afrag[mi], false, bfrag[ni], (short)0, acc[mi][ni], false, false);
    __syncthreads();
  }

  // ---- epilogue (C/D layout: row = v + 8*(lane>=16), col = lane&15) ----
  const int hgrp = lane >> 4;
  const int l16  = lane & 15;
#pragma unroll
  for (int mi = 0; mi < 4; ++mi) {
#pragma unroll
    for (int ni = 0; ni < 2; ++ni) {
      int col = bn + wn * 32 + ni * 16 + l16;
      if (col >= Nc) continue;
#pragma unroll
      for (int v = 0; v < 8; ++v) {
        int row = bm + wm * 64 + mi * 16 + v + 8 * hgrp;
        long long off = (long long)row * ldc + col;
        float val = acc[mi][ni][v];
        if (bias) val += bias[z * sBias + col];
        if (accumulate) val += Cb[off];
        if (act == 1)      val = 1.0f / (1.0f + __expf(-val));   // sigmoid
        else if (act == 2) val = fmaxf(val, 0.0f);               // relu
        if (outBf16) Cb16[off] = (__bf16)val;
        else         Cb[off]   = val;
      }
    }
  }
}

// f32 -> bf16 elementwise (n multiple of 4)
__global__ __launch_bounds__(256) void cvt_bf16_k(const float* __restrict__ src,
    __bf16* __restrict__ dst, long long n)
{
  long long i = ((long long)blockIdx.x * 256 + threadIdx.x) * 4;
  if (i >= n) return;
  float4 f = *(const float4*)(src + i);
  bf16x4 o; o[0] = (__bf16)f.x; o[1] = (__bf16)f.y; o[2] = (__bf16)f.z; o[3] = (__bf16)f.w;
  *(bf16x4*)(dst + i) = o;
}

// per-node 64x64 transpose + cvt: dst[n][o][i] = src[n*nodeStride + i*64 + o]
__global__ __launch_bounds__(256) void cvt_nodeT_k(const float* __restrict__ src,
    long long nodeStride, __bf16* __restrict__ dst)
{
  __shared__ float t[64][65];
  int n = blockIdx.x, tid = threadIdx.x;
  const float* S = src + (long long)n * nodeStride;
#pragma unroll
  for (int j = 0; j < 16; ++j) {
    int idx = tid + j * 256; int i = idx >> 6, o = idx & 63;
    t[i][o] = S[i * 64 + o];
  }
  __syncthreads();
  __bf16* D = dst + (long long)n * 4096;
#pragma unroll
  for (int j = 0; j < 16; ++j) {
    int idx = tid + j * 256; int o = idx >> 6, i = idx & 63;
    D[o * 64 + i] = (__bf16)t[i][o];
  }
}

// LayerNorm(H) + exact GELU, per row; writes bf16 (GEMM operand)
__global__ __launch_bounds__(256) void ln_gelu_k(const float* __restrict__ X,
    const float* __restrict__ g, const float* __restrict__ b,
    __bf16* __restrict__ Y, int Hn)
{
  __shared__ float s1[256], s2[256];
  int row = blockIdx.x, tid = threadIdx.x;
  const float* x = X + (long long)row * Hn;
  float sum = 0.f, sq = 0.f;
  for (int j = tid; j < Hn; j += 256) { float v = x[j]; sum += v; sq += v * v; }
  s1[tid] = sum; s2[tid] = sq; __syncthreads();
  for (int s = 128; s > 0; s >>= 1) {
    if (tid < s) { s1[tid] += s1[tid + s]; s2[tid] += s2[tid + s]; }
    __syncthreads();
  }
  float mu  = s1[0] / Hn;
  float var = s2[0] / Hn - mu * mu;
  float rs  = rsqrtf(var + 1e-5f);
  __bf16* y = Y + (long long)row * Hn;
  for (int j = tid; j < Hn; j += 256) {
    float v = (x[j] - mu) * rs * g[j] + b[j];
    y[j] = (__bf16)(0.5f * v * (1.0f + erff(v * 0.70710678118654752f)));
  }
}

// adjacency: sigmoid(edge_w*1.5)*mask, row-normalized. <<<1,64>>>
__global__ void adj_k(const float* __restrict__ ew, const float* __restrict__ mask,
                      float* __restrict__ adj)
{
  int r = threadIdx.x;
  float tmp[64]; float s = 0.f;
  for (int c = 0; c < 64; ++c) {
    float a = (1.0f / (1.0f + __expf(-ew[r * 64 + c] * 1.5f))) * mask[r * 64 + c];
    tmp[c] = a; s += a;
  }
  float inv = 1.0f / fmaxf(s, 1e-6f);
  for (int c = 0; c < 64; ++c) adj[r * 64 + c] = tmp[c] * inv;
}

// x_agg[b] = adj @ x_embed[b]; writes bf16 (consumed only as GEMM A)
__global__ __launch_bounds__(256) void xagg_k(const float* __restrict__ adj,
    const float* __restrict__ xe, __bf16* __restrict__ xa)
{
  __shared__ float sx[64 * 64];
  __shared__ float sa[64 * 64];
  int b = blockIdx.x, tid = threadIdx.x;
  const float* xrow = xe + (long long)b * FLATD;
#pragma unroll
  for (int i = 0; i < 16; ++i) {
    int idx = tid + i * 256;
    sx[idx] = xrow[idx];
    sa[idx] = adj[idx];
  }
  __syncthreads();
  __bf16* out = xa + (long long)b * FLATD;
#pragma unroll
  for (int i = 0; i < 16; ++i) {
    int idx = tid + i * 256;
    int n = idx >> 6, d = idx & 63;
    float a = 0.f;
#pragma unroll 8
    for (int m = 0; m < 64; ++m) a += sa[n * 64 + m] * sx[m * 64 + d];
    out[idx] = (__bf16)a;
  }
}

// spectral norm: Wn[n] = Wslow[n] / (u^T W W^T u + 1e-8); bf16 out ([o][i])
__global__ void specnorm_k(const float* __restrict__ Ws, const float* __restrict__ u,
                           __bf16* __restrict__ Wn)
{
  __shared__ float v[64];
  __shared__ float red[64];
  __shared__ float sSig;
  int n = blockIdx.x, t = threadIdx.x;
  const float* W  = Ws + (long long)n * DD * DD;  // [o][i]
  const float* un = u + n * DD;
  float acc = 0.f;
  for (int o = 0; o < 64; ++o) acc += W[o * 64 + t] * un[o];   // v[i]
  v[t] = acc; __syncthreads();
  float wv = 0.f;
  for (int i = 0; i < 64; ++i) wv += W[t * 64 + i] * v[i];     // (Wv)[o]
  red[t] = un[t] * wv; __syncthreads();
  for (int s = 32; s > 0; s >>= 1) { if (t < s) red[t] += red[t + s]; __syncthreads(); }
  if (t == 0) sSig = red[0];
  __syncthreads();
  float inv = 1.0f / (sSig + 1e-8f);
  __bf16* Wno = Wn + (long long)n * DD * DD;
  for (int i = 0; i < 64; ++i) Wno[t * 64 + i] = (__bf16)(W[t * 64 + i] * inv);
}

// fold h_prev half of Wf into a per-node bias: fb[n][o] = bf + sum_i hp[i]*Wf[n][D+i][o]
__global__ void fbias_k(const float* __restrict__ Wf, const float* __restrict__ bfv,
                        const float* __restrict__ hprev, float* __restrict__ fb)
{
  int n = blockIdx.x, o = threadIdx.x;
  const float* W  = Wf + (long long)n * 2 * DD * DD;
  const float* hp = hprev + n * DD;
  float s = bfv[n * DD + o];
  for (int i = 0; i < 64; ++i) s += hp[i] * W[(DD + i) * DD + o];
  fb[n * DD + o] = s;
}

// h_raw = i_t*(h_slow+h_fast) + f_t*h_prev  (f32 + bf16 mirror for GEMM A)
__global__ __launch_bounds__(256) void hraw_k(const float* __restrict__ it,
    const float* __restrict__ ft, const float* __restrict__ hs,
    const float* __restrict__ hprev, float* __restrict__ hr,
    __bf16* __restrict__ hrb, int total)
{
  int idx = blockIdx.x * 256 + threadIdx.x;
  if (idx >= total) return;
  int rest = idx & (FLATD - 1);
  float v = it[idx] * hs[idx] + ft[idx] * hprev[rest];
  hr[idx]  = v;
  hrb[idx] = (__bf16)v;
}

// per-(b,n) LN over D of o_t*tanh(h_raw); f32 + bf16 mirror
__global__ void cellln_k(const float* __restrict__ ot, const float* __restrict__ hr,
    const float* __restrict__ g, const float* __restrict__ b,
    float* __restrict__ xf, __bf16* __restrict__ xfb)
{
  __shared__ float s1[64], s2[64];
  long long blk = blockIdx.x;           // b*N + n
  int t = threadIdx.x;
  long long base = blk * 64;
  int nd = (int)(blk & 63) * 64 + t;
  float v = ot[base + t] * tanhf(hr[base + t]);
  s1[t] = v; s2[t] = v * v; __syncthreads();
  for (int s = 32; s > 0; s >>= 1) {
    if (t < s) { s1[t] += s1[t + s]; s2[t] += s2[t + s]; }
    __syncthreads();
  }
  float mu  = s1[0] * (1.0f / 64.0f);
  float var = s2[0] * (1.0f / 64.0f) - mu * mu;
  float rs  = rsqrtf(var + 1e-5f);
  float o = (v - mu) * rs * g[nd] + b[nd];
  xf[base + t]  = o;
  xfb[base + t] = (__bf16)o;
}

// top-5 of sim row, x_flat += 0.1*mean(mem[idx]); mirror into d_out + bf16
__global__ __launch_bounds__(256) void topk_k(const float* __restrict__ sim,
    const float* __restrict__ mem, float* __restrict__ xf,
    float* __restrict__ xout, __bf16* __restrict__ xfb)
{
  __shared__ float s[64];
  __shared__ int idxs[5];
  int b = blockIdx.x, t = threadIdx.x;
  if (t < 64) s[t] = sim[(long long)b * 64 + t];
  __syncthreads();
  if (t == 0) {
    float vals[64];
    for (int i = 0; i < 64; ++i) vals[i] = s[i];
    for (int k = 0; k < 5; ++k) {
      int bi = 0; float bv = -3.4e38f;
      for (int i = 0; i < 64; ++i) if (vals[i] > bv) { bv = vals[i]; bi = i; }
      idxs[k] = bi; vals[bi] = -3.4e38f;
    }
  }
  __syncthreads();
  int i0 = idxs[0], i1 = idxs[1], i2 = idxs[2], i3 = idxs[3], i4 = idxs[4];
  long long base = (long long)b * FLATD;
  for (int j = t; j < FLATD; j += 256) {
    float r = (mem[(long long)i0 * FLATD + j] + mem[(long long)i1 * FLATD + j] +
               mem[(long long)i2 * FLATD + j] + mem[(long long)i3 * FLATD + j] +
               mem[(long long)i4 * FLATD + j]) * 0.2f;
    float nv = xf[base + j] + 0.1f * r;
    xf[base + j]   = nv;
    xout[base + j] = nv;
    xfb[base + j]  = (__bf16)nv;
  }
}

// pre-average the 3 ensemble heads (mean is linear); bf16 weights out
__global__ void wavg_k(const float* __restrict__ Wh, __bf16* __restrict__ Wavg, int total)
{
  int i = blockIdx.x * 256 + threadIdx.x;
  if (i < total) Wavg[i] = (__bf16)((Wh[i] + Wh[i + total] + Wh[i + 2 * total]) * (1.0f / 3.0f));
}
__global__ void bavg_k(const float* __restrict__ bh, float* __restrict__ bavg, int C)
{
  int i = blockIdx.x * 256 + threadIdx.x;
  if (i < C) bavg[i] = (bh[i] + bh[i + C] + bh[i + 2 * C]) * (1.0f / 3.0f);
}

// ---------------------------------------------------------------------------
extern "C" void kernel_launch(void* const* d_in, const int* in_sizes, int n_in,
                              void* d_out, int out_size, void* d_ws, size_t ws_size,
                              hipStream_t stream)
{
  const float* x      = (const float*)d_in[0];
  const float* W1     = (const float*)d_in[1];
  const float* b1     = (const float*)d_in[2];
  const float* g1     = (const float*)d_in[3];
  const float* beta1  = (const float*)d_in[4];
  const float* W2     = (const float*)d_in[5];
  const float* b2     = (const float*)d_in[6];
  const float* edge_w = (const float*)d_in[7];
  const float* maskp  = (const float*)d_in[8];
  const float* Wi     = (const float*)d_in[9];
  const float* bi     = (const float*)d_in[10];
  const float* Wf     = (const float*)d_in[11];
  const float* bf_in  = (const float*)d_in[12];
  const float* Wslow  = (const float*)d_in[13];
  const float* bslow  = (const float*)d_in[14];
  const float* u      = (const float*)d_in[15];
  const float* Wfast  = (const float*)d_in[16];
  const float* Wo     = (const float*)d_in[17];
  const float* bo     = (const float*)d_in[18];
  const float* gln    = (const float*)d_in[19];
  const float* bln    = (const float*)d_in[20];
  const float* h_prev = (const float*)d_in[21];
  const float* memory = (const float*)d_in[22];
  const float* Wh     = (const float*)d_in[23];
  const float* bh     = (const float*)d_in[24];
  const float* Ws1    = (const float*)d_in[25];
  const float* bs1    = (const float*)d_in[26];
  const float* Ws2    = (const float*)d_in[27];
  const float* bs2    = (const float*)d_in[28];

  const long long SZ_BH = (long long)BSZ * HDIM;
  const long long SZ_BF = (long long)BSZ * FLATD;
  const long long SZ_ND = (long long)NN * DD * DD;

  float* ws = (float*)d_ws;
  long long off = 0;
  float* h    = ws + off; off += SZ_BH;            // pre-LN [B,H] f32
  float* xe   = ws + off; off += SZ_BF;            // x_embed f32
  float* it   = ws + off; off += SZ_BF;
  float* ft   = ws + off; off += SZ_BF;
  float* hs   = ws + off; off += SZ_BF;            // h_slow + h_fast
  float* hr   = ws + off; off += SZ_BF;            // h_raw f32
  float* ot   = ws + off; off += SZ_BF;
  float* xf   = ws + off; off += SZ_BF;            // x_flat f32
  float* simb = ws + off; off += (long long)BSZ * CAPM;
  float* adjb = ws + off; off += NN * NN;
  float* fb   = ws + off; off += NN * DD;
  float* bavg = ws + off; off += 1024;

  __bf16* wsb = (__bf16*)(ws + off);
  long long ob = 0;
  __bf16* x_b    = wsb + ob; ob += (long long)BSZ * FDIM;
  __bf16* h_b    = wsb + ob; ob += SZ_BH;
  __bf16* xa_b   = wsb + ob; ob += SZ_BF;
  __bf16* hr_b   = wsb + ob; ob += SZ_BF;
  __bf16* xf_b   = wsb + ob; ob += SZ_BF;
  __bf16* hp2_b  = wsb + ob; ob += SZ_BH;
  __bf16* W1b    = wsb + ob; ob += (long long)HDIM * FDIM;
  __bf16* W2b    = wsb + ob; ob += (long long)FLATD * HDIM;
  __bf16* Wib    = wsb + ob; ob += SZ_ND;
  __bf16* Wfb    = wsb + ob; ob += SZ_ND;
  __bf16* Wnb    = wsb + ob; ob += SZ_ND;
  __bf16* Wfastb = wsb + ob; ob += SZ_ND;
  __bf16* Wob    = wsb + ob; ob += SZ_ND;
  __bf16* memb   = wsb + ob; ob += (long long)CAPM * FLATD;
  __bf16* Wavgb  = wsb + ob; ob += (long long)CCLS * FLATD;
  __bf16* Ws1b   = wsb + ob; ob += (long long)HDIM * FLATD;
  __bf16* Ws2b   = wsb + ob; ob += (long long)8 * HDIM;

  float* out_logits = (float*)d_out;                               // [B,C]
  float* out_proj   = out_logits + (long long)BSZ * CCLS;          // [B,P]
  float* out_xflat  = out_proj + (long long)BSZ * 8;               // [B,FLAT]

  auto cvt = [&](const float* s, __bf16* d, long long n) {
    cvt_bf16_k<<<dim3((unsigned)((n / 4 + 255) / 256)), 256, 0, stream>>>(s, d, n);
  };
  auto gemm = [&](const __bf16* A, long long lda, long long sA,
                  const __bf16* Bm, long long ldb, long long sB,
                  const float* bias, long long sBias,
                  float* C, long long ldc, long long sC,
                  int M, int Nc, int K, int accum, int act, int outBf16, int batch) {
    dim3 g(M / 128, (Nc + 127) / 128, batch);
    gemm_bf16_wmma<<<g, 256, 0, stream>>>(A, lda, sA, Bm, ldb, sB, bias, sBias,
                                          C, ldc, sC, M, Nc, K, accum, act, outBf16);
  };

  // 0) one-time bf16 conversions (weights) + input
  cvt(x, x_b, (long long)BSZ * FDIM);
  cvt(W1, W1b, (long long)HDIM * FDIM);
  cvt(W2, W2b, (long long)FLATD * HDIM);
  cvt(Wfast, Wfastb, SZ_ND);                                  // already [n][o][i]
  cvt(memory, memb, (long long)CAPM * FLATD);
  cvt(Ws1, Ws1b, (long long)HDIM * FLATD);
  cvt(Ws2, Ws2b, (long long)8 * HDIM);
  cvt_nodeT_k<<<NN, 256, 0, stream>>>(Wi, (long long)DD * DD, Wib);        // [n][i][o]->[n][o][i]
  cvt_nodeT_k<<<NN, 256, 0, stream>>>(Wf, (long long)2 * DD * DD, Wfb);    // first D rows
  cvt_nodeT_k<<<NN, 256, 0, stream>>>(Wo, (long long)DD * DD, Wob);
  wavg_k<<<(CCLS * FLATD + 255) / 256, 256, 0, stream>>>(Wh, Wavgb, CCLS * FLATD);
  bavg_k<<<(CCLS + 255) / 256, 256, 0, stream>>>(bh, bavg, CCLS);

  // 1) h = gelu(LN(x @ W1^T + b1))
  gemm(x_b, FDIM, 0, W1b, FDIM, 0, b1, 0, h, HDIM, 0, BSZ, HDIM, FDIM, 0, 0, 0, 1);
  ln_gelu_k<<<BSZ, 256, 0, stream>>>(h, g1, beta1, h_b, HDIM);

  // 2) x_embed = h @ W2^T + b2
  gemm(h_b, HDIM, 0, W2b, HDIM, 0, b2, 0, xe, FLATD, 0, BSZ, FLATD, HDIM, 0, 0, 0, 1);

  // 3) adjacency + aggregation (bf16 out)
  adj_k<<<1, 64, 0, stream>>>(edge_w, maskp, adjb);
  xagg_k<<<BSZ, 256, 0, stream>>>(adjb, xe, xa_b);

  // 4) spectral norm (bf16 out) + folded forget-gate bias
  specnorm_k<<<NN, 64, 0, stream>>>(Wslow, u, Wnb);
  fbias_k<<<NN, 64, 0, stream>>>(Wf, bf_in, h_prev, fb);

  // 5) per-node batched GEMMs (batch = node via blockIdx.z); all A@B^T now
  gemm(xa_b, FLATD, DD, Wib, DD, SZ_ND / NN, bi, DD,
       it, FLATD, DD, BSZ, DD, DD, 0, 1, 0, NN);                    // i_t = sigmoid
  gemm(xa_b, FLATD, DD, Wfb, DD, SZ_ND / NN, fb, DD,
       ft, FLATD, DD, BSZ, DD, DD, 0, 1, 0, NN);                    // f_t = sigmoid
  gemm(xa_b, FLATD, DD, Wnb, DD, SZ_ND / NN, bslow, DD,
       hs, FLATD, DD, BSZ, DD, DD, 0, 0, 0, NN);                    // h_slow
  gemm(xa_b, FLATD, DD, Wfastb, DD, SZ_ND / NN, (const float*)nullptr, 0,
       hs, FLATD, DD, BSZ, DD, DD, 1, 0, 0, NN);                    // += h_fast
  hraw_k<<<(int)(SZ_BF / 256), 256, 0, stream>>>(it, ft, hs, h_prev, hr, hr_b, (int)SZ_BF);
  gemm(hr_b, FLATD, DD, Wob, DD, SZ_ND / NN, bo, DD,
       ot, FLATD, DD, BSZ, DD, DD, 0, 1, 0, NN);                    // o_t = sigmoid

  // 6) per-(b,n) LayerNorm -> x_flat (f32 + bf16)
  cellln_k<<<BSZ * NN, 64, 0, stream>>>(ot, hr, gln, bln, xf, xf_b);

  // 7) episodic retrieval
  gemm(xf_b, FLATD, 0, memb, FLATD, 0, (const float*)nullptr, 0,
       simb, CAPM, 0, BSZ, CAPM, FLATD, 0, 0, 0, 1);
  topk_k<<<BSZ, 256, 0, stream>>>(simb, memory, xf, out_xflat, xf_b);

  // 8) averaged ensemble head
  gemm(xf_b, FLATD, 0, Wavgb, FLATD, 0, bavg, 0,
       out_logits, CCLS, 0, BSZ, CCLS, FLATD, 0, 0, 0, 1);

  // 9) supcon projection (hidden stored bf16 straight from epilogue)
  gemm(xf_b, FLATD, 0, Ws1b, FLATD, 0, bs1, 0, (float*)hp2_b, HDIM, 0,
       BSZ, HDIM, FLATD, 0, 2, 1, 1);                               // relu, bf16 out
  gemm(hp2_b, HDIM, 0, Ws2b, HDIM, 0, bs2, 0, out_proj, 8, 0,
       BSZ, 8, HDIM, 0, 0, 0, 1);
}
